// NeuMFHead_35562329211060
// MI455X (gfx1250) — matrix operand order repaired and checked by reading, not compiled, not run on hardware
//
#include <hip/hip_runtime.h>

// ---------------------------------------------------------------------------
// NeuMF head for MI455X (gfx1250): bf16 WMMA with fp32 accumulation.
//   x[B,832] = concat(u_mat[u], i_mat[i])
//   h1 = relu(x @ W1 + b1)   [B,128]   (dominant GEMM -> WMMA bf16)
//   h2 = relu(h1 @ W2 + b2)  [B,64]    (WMMA bf16)
//   out = h2 @ W3 + b3       [B]       (VALU dot + lane reduction)
// ---------------------------------------------------------------------------

namespace {
constexpr int kBatch = 262144;
constexpr int kUD = 64;
constexpr int kID = 768;
constexpr int kX  = 832;            // kUD + kID
constexpr int kH1 = 128;
constexpr int kH2 = 64;

constexpr int kKB1 = kX / 32;       // 26 k-steps for layer 1
constexpr int kKB2 = kH1 / 32;      // 4 k-steps for layer 2

// LDS layout (bytes):
//  W1 bf16 B-fragments: [kb][half][n][16]  -> ((kb*2+half)*128 + n)*32
//  W2 bf16 B-fragments: [kb][half][n][16]  -> ((kb*2+half)*64  + n)*32
//  h1 staging: per wave, 2 tiles of 16x128 bf16 (row-major, row stride 256B)
constexpr int kW1Bytes = kKB1 * 2 * kH1 * 32;   // 212992
constexpr int kW2Bytes = kKB2 * 2 * kH2 * 32;   // 16384
constexpr int kWaves   = 8;
constexpr int kH1PerWave = 2 * 16 * kH1 * 2;    // 8192
constexpr int kLds = kW1Bytes + kW2Bytes + kWaves * kH1PerWave; // 294912 <= 320KB
} // namespace

typedef __attribute__((ext_vector_type(16))) __bf16       v16bf;
typedef __attribute__((ext_vector_type(8)))  float        v8f;
typedef __attribute__((ext_vector_type(4)))  float        v4f;
typedef __attribute__((ext_vector_type(4)))  unsigned int v4u;

// Build a WMMA A-fragment (16x32 bf16, wave32 layout) from 16 contiguous-ish
// fp32 values: elements 0..7 <- p[0..7], elements 8..15 <- p[16..23].
// (lane 0-15: K = c..c+7, c+16..c+23 ; lane 16-31: caller offsets p by +8)
__device__ __forceinline__ v16bf a_frag(const float* __restrict__ p) {
  v4f x0 = *(const v4f*)(p + 0);
  v4f x1 = *(const v4f*)(p + 4);
  v4f x2 = *(const v4f*)(p + 16);
  v4f x3 = *(const v4f*)(p + 20);
  v16bf a;
  a[0]  = (__bf16)x0[0]; a[1]  = (__bf16)x0[1]; a[2]  = (__bf16)x0[2]; a[3]  = (__bf16)x0[3];
  a[4]  = (__bf16)x1[0]; a[5]  = (__bf16)x1[1]; a[6]  = (__bf16)x1[2]; a[7]  = (__bf16)x1[3];
  a[8]  = (__bf16)x2[0]; a[9]  = (__bf16)x2[1]; a[10] = (__bf16)x2[2]; a[11] = (__bf16)x2[3];
  a[12] = (__bf16)x3[0]; a[13] = (__bf16)x3[1]; a[14] = (__bf16)x3[2]; a[15] = (__bf16)x3[3];
  return a;
}

// B-fragment: 32 contiguous bytes (16 packed bf16) from LDS.
__device__ __forceinline__ v16bf b_frag(const char* p) {
  v16bf b;
  ((v4u*)&b)[0] = *(const v4u*)(p);
  ((v4u*)&b)[1] = *(const v4u*)(p + 16);
  return b;
}

__device__ __forceinline__ v8f wmma_bf16(v16bf a, v16bf b, v8f c) {
  return __builtin_amdgcn_wmma_f32_16x16x32_bf16(false, a, false, b, (short)0, c,
                                                 false, false);
}

__global__ __launch_bounds__(256, 1)
void neumf_kernel(const float* __restrict__ u_mat,
                  const float* __restrict__ i_mat,
                  const float* __restrict__ W1,
                  const float* __restrict__ b1,
                  const float* __restrict__ W2,
                  const float* __restrict__ b2,
                  const float* __restrict__ W3,
                  const float* __restrict__ b3,
                  const int*   __restrict__ uIdx,
                  const int*   __restrict__ iIdx,
                  float*       __restrict__ out)
{
  extern __shared__ char smem[];
  __bf16* ldsW1 = (__bf16*)smem;
  __bf16* ldsW2 = (__bf16*)(smem + kW1Bytes);
  char*   ldsH1 = smem + kW1Bytes + kW2Bytes;

  const int tid  = threadIdx.x;
  const int wave = tid >> 5;
  const int lane = tid & 31;
  const int lid  = lane & 15;   // column / row id within 16
  const int hi   = lane >> 4;   // which K-half this lane owns

  // ---- Stage W1/W2 into LDS, converted to bf16 in B-fragment order. ----
  // B layout (16-bit 32x16): lane n (0-15) holds column n, K=0..7,16..23
  // packed; lanes 16-31 hold K=8..15,24..31.  Element (kg,n) -> slot
  // j=(kg&7)|((kg>>1)&8) of block (kb=kg>>5, half=(kg>>3)&1, n).  Reads of W1 are fully coalesced.
  for (int e = tid; e < kX * kH1; e += 256) {
    int kg = e >> 7, n = e & (kH1 - 1);
    int kb = kg >> 5, half = (kg >> 3) & 1, j = (kg & 7) | ((kg >> 1) & 8);
    ldsW1[(((kb * 2 + half) * kH1 + n) << 4) + j] = (__bf16)W1[e];
  }
  for (int e = tid; e < kH1 * kH2; e += 256) {
    int kg = e >> 6, n = e & (kH2 - 1);
    int kb = kg >> 5, half = (kg >> 3) & 1, j = (kg & 7) | ((kg >> 1) & 8);
    ldsW2[(((kb * 2 + half) * kH2 + n) << 4) + j] = (__bf16)W2[e];
  }
  __syncthreads();

  // Per-lane biases / W3 column values (this lane's N = n*16 + lid).
  float b1v[8], b2v[4], w3v[4];
#pragma unroll
  for (int n = 0; n < 8; ++n) b1v[n] = b1[n * 16 + lid];
#pragma unroll
  for (int n = 0; n < 4; ++n) b2v[n] = b2[n * 16 + lid];
#pragma unroll
  for (int n = 0; n < 4; ++n) w3v[n] = W3[n * 16 + lid];
  const float b3s = b3[0];

  // Per-lane base pointers into the fragment-packed weight regions.
  const char* bw1 = (const char*)smem + hi * (kH1 * 32) + lid * 32;            // + kb*8192 + nt*512
  const char* bw2 = (const char*)smem + kW1Bytes + hi * (kH2 * 32) + lid * 32; // + kb*4096 + nt*512
  char* h1w = ldsH1 + wave * kH1PerWave;

  const int nChunk = kBatch / 32;   // each wave-chunk = 32 rows = 2 M-tiles
  for (int c = blockIdx.x * kWaves + wave; c < nChunk; c += gridDim.x * kWaves) {
    // Gather row pointers (lanes L and L+16 own the same row: same lid).
    const float* up[2];
    const float* ip[2];
#pragma unroll
    for (int t = 0; t < 2; ++t) {
      int row = c * 32 + t * 16 + lid;
      up[t] = u_mat + (size_t)uIdx[row] * kUD;
      ip[t] = i_mat + (size_t)iIdx[row] * kID;
    }

    // ---------------- Layer 1: x @ W1 (26 k-steps, 8 n-tiles) -------------
    v8f acc[2][8];
#pragma unroll
    for (int t = 0; t < 2; ++t)
#pragma unroll
      for (int n = 0; n < 8; ++n) acc[t][n] = {};

#pragma unroll
    for (int kb = 0; kb < 2; ++kb) {   // user features: global cols 0..63
      v16bf a0 = a_frag(up[0] + kb * 32 + hi * 8);
      v16bf a1 = a_frag(up[1] + kb * 32 + hi * 8);
      const char* bp = bw1 + kb * (2 * kH1 * 32);
#pragma unroll
      for (int n = 0; n < 8; ++n) {
        v16bf bf = b_frag(bp + n * 512);
        acc[0][n] = wmma_bf16(a0, bf, acc[0][n]);
        acc[1][n] = wmma_bf16(a1, bf, acc[1][n]);
      }
    }
    for (int kb = 0; kb < 24; ++kb) {  // item features: global cols 64..831
      v16bf a0 = a_frag(ip[0] + kb * 32 + hi * 8);
      v16bf a1 = a_frag(ip[1] + kb * 32 + hi * 8);
      const char* bp = bw1 + (kb + 2) * (2 * kH1 * 32);
#pragma unroll
      for (int n = 0; n < 8; ++n) {
        v16bf bf = b_frag(bp + n * 512);
        acc[0][n] = wmma_bf16(a0, bf, acc[0][n]);
        acc[1][n] = wmma_bf16(a1, bf, acc[1][n]);
      }
    }

    // Bias + ReLU, stage h1 as bf16 [M][N] per tile (row stride 256B).
    // C layout: lane holds (M = r + hi*8, N = n*16 + lid).
#pragma unroll
    for (int t = 0; t < 2; ++t) {
      char* h1t = h1w + t * 4096;
#pragma unroll
      for (int n = 0; n < 8; ++n) {
#pragma unroll
        for (int r = 0; r < 8; ++r) {
          float v = acc[t][n][r] + b1v[n];
          v = v > 0.f ? v : 0.f;
          *(__bf16*)(h1t + (((r + hi * 8) * kH1 + n * 16 + lid) << 1)) = (__bf16)v;
        }
      }
    }

    // ---------------- Layers 2 + 3, per tile ------------------------------
#pragma unroll
    for (int t = 0; t < 2; ++t) {
      const char* h1t = h1w + t * 4096;
      v8f acc2[4];
#pragma unroll
      for (int n = 0; n < 4; ++n) acc2[n] = {};
#pragma unroll
      for (int kb = 0; kb < kKB2; ++kb) {
        // A2 fragment: row M=lid, K = kb*32 + (hi?8:0) + {0..7,16..23}
        const char* ap = h1t + lid * (kH1 * 2) + ((kb * 32 + hi * 8) << 1);
        v16bf a2;
        ((v4u*)&a2)[0] = *(const v4u*)(ap);
        ((v4u*)&a2)[1] = *(const v4u*)(ap + 32);
        const char* bp = bw2 + kb * (2 * kH2 * 32);
#pragma unroll
        for (int n = 0; n < 4; ++n) {
          v16bf bf = b_frag(bp + n * 512);
          acc2[n] = wmma_bf16(a2, bf, acc2[n]);
        }
      }
      // h2 = relu(acc2 + b2); out = h2 . W3 + b3  (reduce across 16 lanes)
      float s[8];
#pragma unroll
      for (int r = 0; r < 8; ++r) {
        float v = 0.f;
#pragma unroll
        for (int n = 0; n < 4; ++n) {
          float h = acc2[n][r] + b2v[n];
          h = h > 0.f ? h : 0.f;
          v += h * w3v[n];
        }
#pragma unroll
        for (int m = 1; m < 16; m <<= 1) v += __shfl_xor(v, m, 32);
        s[r] = v + b3s;
      }
      if (lid == 0) {   // lane 0 -> M=0..7, lane 16 -> M=8..15
        int rb = c * 32 + t * 16 + hi * 8;
#pragma unroll
        for (int r = 0; r < 8; ++r) out[rb + r] = s[r];
      }
    }
  }
}

extern "C" void kernel_launch(void* const* d_in, const int* in_sizes, int n_in,
                              void* d_out, int out_size, void* d_ws, size_t ws_size,
                              hipStream_t stream) {
  (void)in_sizes; (void)n_in; (void)out_size; (void)d_ws; (void)ws_size;
  const float* u_mat = (const float*)d_in[0];
  const float* i_mat = (const float*)d_in[1];
  const float* W1    = (const float*)d_in[2];
  const float* b1    = (const float*)d_in[3];
  const float* W2    = (const float*)d_in[4];
  const float* b2    = (const float*)d_in[5];
  const float* W3    = (const float*)d_in[6];
  const float* b3    = (const float*)d_in[7];
  const int*   u     = (const int*)d_in[8];
  const int*   i     = (const int*)d_in[9];
  float* out = (float*)d_out;

  hipFuncSetAttribute((const void*)neumf_kernel,
                      hipFuncAttributeMaxDynamicSharedMemorySize, kLds);
  neumf_kernel<<<dim3(512), dim3(256), kLds, stream>>>(
      u_mat, i_mat, W1, b1, W2, b2, W3, b3, u, i, out);
}
